// BayesianExpectationTransformerLayer_73108933312702
// MI455X (gfx1250) — matrix operand order repaired and checked by reading, not compiled, not run on hardware
//
#include <hip/hip_runtime.h>
#include <hip/hip_bf16.h>

// ---------------------------------------------------------------------------
// BayesianExpectationTransformerLayer on MI455X (gfx1250, wave32, WMMA)
//
// Identity: softmax attention is permutation-equivariant, so every permuted
// pass equals the standard pass after inverse permutation:
//     combined = [(1-w) + w*vrw] * out_proj(attention(x))
// Strategy: convert operands to f16 ONCE, then run the whole MHA through
// v_wmma_f32_16x16x32_f16 with f32 accumulation and f32 softmax.  All inner
// loops load pure f16 (b128 vector loads, no cvt), V is stored transposed so
// the P*V fragments are contiguous.
// ---------------------------------------------------------------------------

typedef __attribute__((ext_vector_type(16))) _Float16 v16h;
typedef __attribute__((ext_vector_type(8)))  _Float16 v8h;
typedef __attribute__((ext_vector_type(4)))  _Float16 v4h;
typedef __attribute__((ext_vector_type(8)))  float    v8f;
typedef __attribute__((ext_vector_type(4)))  float    v4f;

#define WMMA_F16(a, b, c) \
    __builtin_amdgcn_wmma_f32_16x16x32_f16(false, (a), false, (b), (short)0, (c), false, false)

static constexpr int BB = 2;
static constexpr int S  = 1024;
static constexpr int D  = 1024;
static constexpr int H  = 16;
static constexpr int HD = 64;
static constexpr int M  = BB * S;                      // 2048 flattened (b,s)
static constexpr float LOG_S_OVER_S = 0.0067690154f;   // ln(1024)/1024

// Build a v16h from two adjacent 16-byte chunks (contiguous A-fragment layout).
__device__ __forceinline__ v16h load_a_chunks(const _Float16* p) {
    v8h lo = *(const v8h*)(p);
    v8h hi = *(const v8h*)(p + 8);
    return __builtin_shufflevector(lo, hi, 0,1,2,3,4,5,6,7,8,9,10,11,12,13,14,15);
}

// ---------------------------------------------------------------------------
// Kernel 0: f32 -> f16 convert (one pass; 4 elements per thread)
// ---------------------------------------------------------------------------
__global__ void cvt_f32_to_f16_kernel(const float* __restrict__ src,
                                      _Float16* __restrict__ dst, int n)
{
    const int i = (blockIdx.x * blockDim.x + threadIdx.x) * 4;
    if (i >= n) return;
    v4f s = *(const v4f*)(src + i);
    v4h d;
    d[0] = (_Float16)s[0]; d[1] = (_Float16)s[1];
    d[2] = (_Float16)s[2]; d[3] = (_Float16)s[3];
    *(v4h*)(dst + i) = d;
}

// ---------------------------------------------------------------------------
// Kernel 1: fused Q/K/V projection, all-f16 inner loop.
// Per wave: 32(M) x 64(N) tile, K-step 32 -> 8 wmma per iteration.
// Q written pre-scaled by 1/8 head-major [B,H,S,HD];
// K written head-major [B,H,S,HD];  V written TRANSPOSED [B,H,HD,S].
// grid = (D/64, M/32, 3)   block = 32
// ---------------------------------------------------------------------------
__global__ void qkv_proj_kernel(const _Float16* __restrict__ xh,
                                const _Float16* __restrict__ Wqh,
                                const _Float16* __restrict__ Wkh,
                                const _Float16* __restrict__ Wvh,
                                const float* __restrict__ bq,
                                const float* __restrict__ bk,
                                const float* __restrict__ bv,
                                _Float16* __restrict__ Qh,
                                _Float16* __restrict__ Kh,
                                _Float16* __restrict__ VTh)
{
    const int lane = threadIdx.x & 31;
    const int hi   = lane >> 4;
    const int lo   = lane & 15;
    const int n0   = blockIdx.x * 64;
    const int m0   = blockIdx.y * 32;
    const int sel  = blockIdx.z;

    const _Float16* W = (sel == 0) ? Wqh : (sel == 1) ? Wkh : Wvh;
    const float* bias = (sel == 0) ? bq  : (sel == 1) ? bk  : bv;

    v8f acc[8];
#pragma unroll
    for (int f = 0; f < 8; ++f)
#pragma unroll
        for (int i = 0; i < 8; ++i) acc[f][i] = 0.0f;

    const _Float16* arow0 = xh + (size_t)(m0 + lo) * D;
    const _Float16* arow1 = arow0 + (size_t)16 * D;

    for (int k0 = 0; k0 < D; k0 += 32) {
        if (k0 + 64 < D) {   // gfx1250 global_prefetch of next A tiles
            __builtin_prefetch(arow0 + k0 + 64, 0, 0);
            __builtin_prefetch(arow1 + k0 + 64, 0, 0);
        }
        const v16h a0 = load_a_chunks(arow0 + k0 + 16 * hi);
        const v16h a1 = load_a_chunks(arow1 + k0 + 16 * hi);
#pragma unroll
        for (int f = 0; f < 4; ++f) {
            const v16h bf = *(const v16h*)(W + (size_t)(n0 + f * 16 + lo) * D + k0 + 16 * hi);
            acc[f]     = WMMA_F16(a0, bf, acc[f]);
            acc[4 + f] = WMMA_F16(a1, bf, acc[4 + f]);
        }
    }

    const float mul = (sel == 0) ? 0.125f : 1.0f;   // fold softmax scale into Q
#pragma unroll
    for (int f = 0; f < 4; ++f) {
        const int e  = n0 + f * 16 + lo;
        const int hh = e >> 6;
        const int dd = e & 63;
        const float bs = bias[e];
#pragma unroll
        for (int half = 0; half < 2; ++half) {
#pragma unroll
            for (int v = 0; v < 8; ++v) {
                const int m = m0 + 16 * half + v + 8 * hi;
                const int b = m >> 10;
                const int s = m & 1023;
                const float val = (acc[4 * half + f][v] + bs) * mul;
                if (sel == 2) {
                    VTh[(((size_t)b * H + hh) * HD + dd) * S + s] = (_Float16)val;
                } else {
                    _Float16* outp = (sel == 0) ? Qh : Kh;
                    outp[(((size_t)b * H + hh) * S + s) * HD + dd] = (_Float16)val;
                }
            }
        }
    }
}

// ---------------------------------------------------------------------------
// Kernel 2: flash-style attention per (b,h); one wave owns 16 query rows,
// streams keys in tiles of 32 with online softmax.  All operands f16.
// grid = (S/16, B*H)   block = 32
// ---------------------------------------------------------------------------
__global__ void attn_kernel(const _Float16* __restrict__ Qh,
                            const _Float16* __restrict__ Kh,
                            const _Float16* __restrict__ VTh,
                            _Float16* __restrict__ Oh)
{
    __shared__ float pls[16 * 32];     // P-tile staging: C-layout -> A-layout

    const int lane = threadIdx.x & 31;
    const int hi   = lane >> 4;
    const int lo   = lane & 15;
    const int bh   = blockIdx.y;
    const int q0   = blockIdx.x * 16;

    const _Float16* Q  = Qh  + (size_t)bh * S * HD;
    const _Float16* K  = Kh  + (size_t)bh * S * HD;
    const _Float16* VT = VTh + (size_t)bh * HD * S;

    // Q A-fragments (16 x 64 = two 16x32 frags); scale already folded in.
    const _Float16* qp = Q + (size_t)(q0 + lo) * HD + 16 * hi;
    const v16h qa0 = load_a_chunks(qp);
    const v16h qa1 = load_a_chunks(qp + 32);

    float mrow[8], lrow[8];
    v8f o[4];
#pragma unroll
    for (int v = 0; v < 8; ++v) { mrow[v] = -1e30f; lrow[v] = 0.0f; }
#pragma unroll
    for (int f = 0; f < 4; ++f)
#pragma unroll
        for (int i = 0; i < 8; ++i) o[f][i] = 0.0f;

    for (int kt = 0; kt < S; kt += 32) {
        // ---- scores: two 16x16 f32 fragments for keys kt..kt+31
        v8f s0, s1;
#pragma unroll
        for (int i = 0; i < 8; ++i) { s0[i] = 0.0f; s1[i] = 0.0f; }
        {
            const _Float16* kp0 = K + (size_t)(kt + lo) * HD + 16 * hi;  // keys kt..+15
            const _Float16* kp1 = kp0 + 16 * HD;                          // keys kt+16..+31
            const v16h b0 = *(const v16h*)(kp0);        // d 0..31
            const v16h b1 = *(const v16h*)(kp0 + 32);   // d 32..63
            const v16h b2 = *(const v16h*)(kp1);
            const v16h b3 = *(const v16h*)(kp1 + 32);
            s0 = WMMA_F16(qa0, b0, s0);
            s0 = WMMA_F16(qa1, b1, s0);
            s1 = WMMA_F16(qa0, b2, s1);
            s1 = WMMA_F16(qa1, b3, s1);
        }

        // ---- online softmax (row stats across the 16-lane column stripe)
        float pr0[8], pr1[8];
#pragma unroll
        for (int v = 0; v < 8; ++v) {
            float mx = fmaxf(s0[v], s1[v]);
#pragma unroll
            for (int off = 8; off >= 1; off >>= 1)
                mx = fmaxf(mx, __shfl_xor(mx, off, 16));
            const float mnew = fmaxf(mrow[v], mx);
            const float corr = __expf(mrow[v] - mnew);
            const float p0   = __expf(s0[v] - mnew);
            const float p1   = __expf(s1[v] - mnew);
            float sum = p0 + p1;
#pragma unroll
            for (int off = 8; off >= 1; off >>= 1)
                sum += __shfl_xor(sum, off, 16);
            lrow[v] = lrow[v] * corr + sum;
            mrow[v] = mnew;
            pr0[v] = p0; pr1[v] = p1;
            o[0][v] *= corr; o[1][v] *= corr; o[2][v] *= corr; o[3][v] *= corr;
        }

        // ---- reshape P: C-layout f32 -> A-layout f16 via LDS
        __syncthreads();
#pragma unroll
        for (int v = 0; v < 8; ++v) {
            const int r = v + 8 * hi;
            pls[r * 32 + lo]      = pr0[v];
            pls[r * 32 + 16 + lo] = pr1[v];
        }
        __syncthreads();
        v16h pa;
#pragma unroll
        for (int h = 0; h < 16; ++h) {
            const int kl = h + 16 * hi;
            pa[h] = (_Float16)pls[lo * 32 + kl];
        }

        // ---- O += P * V : V^T rows are contiguous in key -> pure b128 loads
#pragma unroll
        for (int f = 0; f < 4; ++f) {
            const v16h bf = *(const v16h*)(VT + (size_t)(f * 16 + lo) * S + kt + 16 * hi);
            o[f] = WMMA_F16(pa, bf, o[f]);
        }
    }

    // ---- normalize and store f16 head-major [B,H,S,HD]
#pragma unroll
    for (int v = 0; v < 8; ++v) {
        const float inv = 1.0f / lrow[v];
        const int srow  = q0 + v + 8 * hi;
        _Float16* op = Oh + ((size_t)bh * S + srow) * HD;
#pragma unroll
        for (int f = 0; f < 4; ++f) op[f * 16 + lo] = (_Float16)(o[f][v] * inv);
    }
}

// ---------------------------------------------------------------------------
// Kernel 3: output projection + Bayesian blend, all-f16 inner loop.
// Per wave: 32(M) x 64(N) tile.
// out = [(1-w) + w*vrw] * (attn @ Wo^T + bo),  w = clip(law*ln(S)/S, .01, 1)
// grid = (D/64, M/32)   block = 32
// ---------------------------------------------------------------------------
__global__ void out_proj_kernel(const _Float16* __restrict__ A,   // [B,H,S,HD] f16
                                const _Float16* __restrict__ Woh,
                                const float* __restrict__ bo,
                                const float* __restrict__ vrw,
                                const float* __restrict__ law,
                                float* __restrict__ out)           // [B,S,D] f32
{
    const int lane = threadIdx.x & 31;
    const int hi   = lane >> 4;
    const int lo   = lane & 15;
    const int n0   = blockIdx.x * 64;
    const int m0   = blockIdx.y * 32;

    v8f acc[8];
#pragma unroll
    for (int f = 0; f < 8; ++f)
#pragma unroll
        for (int i = 0; i < 8; ++i) acc[f][i] = 0.0f;

    const int m0a = m0 + lo;        // row for a0
    const int m1a = m0a + 16;       // row for a1
    const int b0r = m0a >> 10, s0r = m0a & 1023;
    const int b1r = m1a >> 10, s1r = m1a & 1023;

    for (int k0 = 0; k0 < D; k0 += 32) {
        // A fragment from head-major f16 buffer; 8-half chunks stay in-head.
        const int kb0 = k0 + 16 * hi;
        const int kb1 = k0 + 16 * hi + 8;
        const _Float16* p00 = A + (((size_t)b0r * H + (kb0 >> 6)) * S + s0r) * HD + (kb0 & 63);
        const _Float16* p01 = A + (((size_t)b0r * H + (kb1 >> 6)) * S + s0r) * HD + (kb1 & 63);
        const _Float16* p10 = A + (((size_t)b1r * H + (kb0 >> 6)) * S + s1r) * HD + (kb0 & 63);
        const _Float16* p11 = A + (((size_t)b1r * H + (kb1 >> 6)) * S + s1r) * HD + (kb1 & 63);
        v16h a0, a1;
        {
            v8h l0 = *(const v8h*)p00, h0 = *(const v8h*)p01;
            v8h l1 = *(const v8h*)p10, h1 = *(const v8h*)p11;
            a0 = __builtin_shufflevector(l0, h0, 0,1,2,3,4,5,6,7,8,9,10,11,12,13,14,15);
            a1 = __builtin_shufflevector(l1, h1, 0,1,2,3,4,5,6,7,8,9,10,11,12,13,14,15);
        }
        if (k0 + 64 < D) {
            __builtin_prefetch(p00 + 64, 0, 0);
            __builtin_prefetch(p10 + 64, 0, 0);
        }
#pragma unroll
        for (int f = 0; f < 4; ++f) {
            const v16h bf = *(const v16h*)(Woh + (size_t)(n0 + f * 16 + lo) * D + k0 + 16 * hi);
            acc[f]     = WMMA_F16(a0, bf, acc[f]);
            acc[4 + f] = WMMA_F16(a1, bf, acc[4 + f]);
        }
    }

    // Blend factor (perm_out == standard_out by permutation equivariance)
    float w = law[0] * LOG_S_OVER_S;
    w = fminf(fmaxf(w, 0.01f), 1.0f);
    const float sc = (1.0f - w) + w * vrw[0];

#pragma unroll
    for (int f = 0; f < 4; ++f) {
        const int e  = n0 + f * 16 + lo;
        const float bs = bo[e];
#pragma unroll
        for (int half = 0; half < 2; ++half) {
#pragma unroll
            for (int v = 0; v < 8; ++v) {
                const int mm = m0 + 16 * half + v + 8 * hi;
                out[(size_t)mm * D + e] = (acc[4 * half + f][v] + bs) * sc;
            }
        }
    }
}

// ---------------------------------------------------------------------------
// Launch
// ---------------------------------------------------------------------------
extern "C" void kernel_launch(void* const* d_in, const int* in_sizes, int n_in,
                              void* d_out, int out_size, void* d_ws, size_t ws_size,
                              hipStream_t stream)
{
    const float* x   = (const float*)d_in[0];
    const float* Wq  = (const float*)d_in[1];
    const float* bq  = (const float*)d_in[2];
    const float* Wk  = (const float*)d_in[3];
    const float* bk  = (const float*)d_in[4];
    const float* Wv  = (const float*)d_in[5];
    const float* bv  = (const float*)d_in[6];
    const float* Wo  = (const float*)d_in[7];
    const float* bo  = (const float*)d_in[8];
    const float* vrw = (const float*)d_in[9];
    const float* law = (const float*)d_in[10];
    // d_in[11] = perms: unused — attention is permutation-equivariant.

    // Workspace layout (f16 tensors), all segments 2 MB-aligned:
    //   xh   : M*D          4 MB
    //   Wqh/Wkh/Wvh/Woh     2 MB each
    //   Qh, Kh              4 MB each (head-major)
    //   VTh                 4 MB     ([B,H,HD,S] transposed)
    //   attn                4 MB     (head-major f16)
    char* ws = (char*)d_ws;
    _Float16* xh   = (_Float16*)(ws);
    _Float16* Wqh  = (_Float16*)(ws + (4) * 1024 * 1024);
    _Float16* Wkh  = (_Float16*)(ws + (6) * 1024 * 1024);
    _Float16* Wvh  = (_Float16*)(ws + (8) * 1024 * 1024);
    _Float16* Woh  = (_Float16*)(ws + (10) * 1024 * 1024);
    _Float16* Qh   = (_Float16*)(ws + (12) * 1024 * 1024);
    _Float16* Kh   = (_Float16*)(ws + (16) * 1024 * 1024);
    _Float16* VTh  = (_Float16*)(ws + (20) * 1024 * 1024);
    _Float16* attn = (_Float16*)(ws + (24) * 1024 * 1024);
    float* out = (float*)d_out;

    const int nX = M * D;     // 2,097,152
    const int nW = D * D;     // 1,048,576
    cvt_f32_to_f16_kernel<<<nX / 1024, 256, 0, stream>>>(x,  xh,  nX);
    cvt_f32_to_f16_kernel<<<nW / 1024, 256, 0, stream>>>(Wq, Wqh, nW);
    cvt_f32_to_f16_kernel<<<nW / 1024, 256, 0, stream>>>(Wk, Wkh, nW);
    cvt_f32_to_f16_kernel<<<nW / 1024, 256, 0, stream>>>(Wv, Wvh, nW);
    cvt_f32_to_f16_kernel<<<nW / 1024, 256, 0, stream>>>(Wo, Woh, nW);

    dim3 gQKV(D / 64, M / 32, 3);     // (16, 64, 3)
    qkv_proj_kernel<<<gQKV, 32, 0, stream>>>(xh, Wqh, Wkh, Wvh, bq, bk, bv,
                                             Qh, Kh, VTh);

    dim3 gATT(S / 16, BB * H);        // (64, 32)
    attn_kernel<<<gATT, 32, 0, stream>>>(Qh, Kh, VTh, attn);

    dim3 gOUT(D / 64, M / 32);        // (16, 64)
    out_proj_kernel<<<gOUT, 32, 0, stream>>>(attn, Woh, bo, vrw, law, out);
}
